// PSMMix3dEmbedding_16862041604437
// MI455X (gfx1250) — matrix operand order repaired and checked by reading, not compiled, not run on hardware
//
#include <hip/hip_runtime.h>

// ---------------------------------------------------------------------------
// PSM Mix3d embedding, CDNA5 (gfx1250, wave32, WMMA bf16).
// dist_sm[i,j] = exp(s_ij)/Z_j (s in [0,1] so no max pass needed);
// 1/Z_j and the j-mask are folded into pos_emb -> GEMM2 is a plain
// bf16 WMMA matmul; GEMM3 (@W_feat^T) fused in the same block with the
// full K=128 pos_feat tile staged through LDS. Double-buffered loads so
// WMMAs overlap L2 latency instead of s_wait_loadcnt 0 before every op.
// ---------------------------------------------------------------------------

constexpr int Ntok = 2048;
constexpr int Dim  = 1024;
constexpr int Kdim = 128;
constexpr int Bsz  = 4;
constexpr int Fdim = 9;

typedef __bf16 bf16;
typedef __attribute__((ext_vector_type(16))) __bf16 v16bf;
typedef __attribute__((ext_vector_type(8)))  float  v8f;

union V16 { v16bf bf; uint4 u4[2]; };

__device__ __forceinline__ v8f wmma_bf16(const v16bf a, const v16bf b, v8f c) {
  return __builtin_amdgcn_wmma_f32_16x16x32_bf16(false, a, false, b, (short)0, c,
                                                 false, false);
}

// --------------------------- W_feat f32 -> bf16 ----------------------------
__global__ __launch_bounds__(256) void wconv_kernel(const float* __restrict__ w,
                                                    bf16* __restrict__ wb) {
  int i = blockIdx.x * 256 + threadIdx.x;   // D*K = 131072 threads
  wb[i] = (bf16)w[i];
}

// ---------------- embedding gather-sum + tuple outputs ---------------------
__global__ __launch_bounds__(256) void embed_kernel(
    const int* __restrict__ token, const int* __restrict__ attr,
    const float* __restrict__ embed, const float* __restrict__ atom,
    float* __restrict__ out) {
  int tokidx = blockIdx.x;                      // b*N + n
  int tid = token[tokidx];
  const float4* er = (const float4*)(embed + (size_t)tid * Dim);
  float4 acc = er[threadIdx.x];
#pragma unroll
  for (int f = 1; f < Fdim; ++f) {
    int a = attr[(size_t)tokidx * Fdim + f];
    float4 v = ((const float4*)(atom + (size_t)a * Dim))[threadIdx.x];
    acc.x += v.x; acc.y += v.y; acc.z += v.z; acc.w += v.w;
  }
  ((float4*)(out + (size_t)tokidx * Dim))[threadIdx.x] = acc;
  if (threadIdx.x == 0) {
    size_t maskOff = (size_t)Bsz * Ntok * Dim;
    out[maskOff + tokidx]              = (tid == 0) ? 1.0f : 0.0f; // padding_mask
    out[maskOff + Bsz * Ntok + tokidx] = (float)tid;               // token_type
  }
}

// ------------- E = exp(s) (bf16 [B,N,N]) + partial column sums -------------
// grid: (j-chunks=8, i-chunks=8, B); block 256 (one thread per j)
__global__ __launch_bounds__(256) void e_kernel(
    const int* __restrict__ token, const float* __restrict__ pos,
    bf16* __restrict__ E, float* __restrict__ Zpart) {
  __shared__ float4 sp[256];
  int b  = blockIdx.z;
  int i0 = blockIdx.y * 256;
  int j  = blockIdx.x * 256 + threadIdx.x;
  {
    int gi = b * Ntok + i0 + threadIdx.x;
    const float* pp = pos + (size_t)gi * 3;
    sp[threadIdx.x] = make_float4(pp[0], pp[1], pp[2],
                                  (token[gi] == 0) ? 1.0f : 0.0f);
  }
  __syncthreads();
  const float* pj = pos + ((size_t)b * Ntok + j) * 3;
  float xj = pj[0], yj = pj[1], zj = pj[2];
  bf16* Ecol = E + (size_t)b * Ntok * Ntok + (size_t)i0 * Ntok + j;
  float zsum = 0.0f;
  for (int ii = 0; ii < 256; ++ii) {
    float4 p = sp[ii];
    float dx = p.x - xj, dy = p.y - yj, dz = p.z - zj;
    float d  = sqrtf(dx * dx + dy * dy + dz * dz);
    float s  = (p.w > 0.5f) ? 0.0f : (1.0f / (d + 1.0f));  // mask on row i
    float e  = __expf(s);
    zsum += e;
    Ecol[(size_t)ii * Ntok] = (bf16)e;
  }
  Zpart[((size_t)b * 8 + blockIdx.y) * Ntok + j] = zsum;   // deterministic
}

// --------- Pt[b][k][j] = mask_j ? 0 : (pos_j . W_pos_k) / Z_j (bf16) -------
// grid: (j-chunks=8, B); block 256
__global__ __launch_bounds__(256) void p_kernel(
    const int* __restrict__ token, const float* __restrict__ pos,
    const float* __restrict__ Wpos, const float* __restrict__ Zpart,
    bf16* __restrict__ Pt) {
  __shared__ float sw[Kdim * 3];
  if (threadIdx.x < Kdim * 3 - 256) sw[256 + threadIdx.x] = Wpos[256 + threadIdx.x];
  sw[threadIdx.x] = Wpos[threadIdx.x];
  __syncthreads();
  int b = blockIdx.y;
  int j = blockIdx.x * 256 + threadIdx.x;
  float Z = 0.0f;
#pragma unroll
  for (int ic = 0; ic < 8; ++ic) Z += Zpart[((size_t)b * 8 + ic) * Ntok + j];
  float rz = 1.0f / Z;
  int gi = b * Ntok + j;
  bool pm = (token[gi] == 0);
  const float* pp = pos + (size_t)gi * 3;
  float px = pp[0], py = pp[1], pz = pp[2];
  bf16* col = Pt + (size_t)b * Kdim * Ntok + j;
  for (int k = 0; k < Kdim; ++k) {
    float v = pm ? 0.0f
                 : (sw[k * 3] * px + sw[k * 3 + 1] * py + sw[k * 3 + 2] * pz) * rz;
    col[(size_t)k * Ntok] = (bf16)v;
  }
}

// --------- fused GEMM2 (E @ Pscaled) + GEMM3 (@ W_feat^T) + add base -------
// grid: B * N/32 = 256 blocks; block 256 = 8 waves.
// wave w: wr=w&1 -> 16-row subtile; wc=w>>1 -> k-quarter (GEMM2) /
// 256-wide d-quarter (GEMM3).
__global__ __launch_bounds__(256) void gemm_kernel(
    const bf16* __restrict__ E, const bf16* __restrict__ Pt,
    const bf16* __restrict__ Wb, const int* __restrict__ token,
    float* __restrict__ out) {
  __shared__ __align__(16) bf16 spf[32 * Kdim];   // pos_feat tile, 8 KB
  __shared__ int smask[32];

  int b  = blockIdx.x >> 6;           // 64 blocks per batch
  int i0 = (blockIdx.x & 63) << 5;    // *32
  int w    = threadIdx.x >> 5;
  int lane = threadIdx.x & 31;
  int half = lane >> 4;
  int ln   = lane & 15;
  int wr = w & 1, wc = w >> 1;        // wc in 0..3

  if (threadIdx.x < 32) smask[threadIdx.x] = token[b * Ntok + i0 + threadIdx.x];

  // ---- GEMM2: pos_feat[i0+wr*16.., wc*32..+31] = sum_j E[i,j] * Pt[k][j]
  // A 16x32 operand, this code's k order: lane (half,m) holds K=half*16..+15 at row m
  // B 32x16 operand, this code's k order: lane (half,n) holds K=half*16..+15 at col n
  const bf16* Ebase = E + ((size_t)b * Ntok + i0 + wr * 16 + ln) * (size_t)Ntok;
  const bf16* Pb = Pt + (size_t)b * Kdim * (size_t)Ntok;
  const bf16* P0 = Pb + (size_t)(wc * 32 + ln) * Ntok + half * 16;       // tile 0
  const bf16* P1 = Pb + (size_t)(wc * 32 + 16 + ln) * Ntok + half * 16;  // tile 1

  v8f acc0 = {}, acc1 = {};
  V16 aA, bA0, bA1, aB, bB0, bB1;
  aA.u4[0]  = *(const uint4*)(Ebase + half * 16);
  aA.u4[1]  = *(const uint4*)(Ebase + half * 16 + 8);
  bA0.u4[0] = *(const uint4*)(P0);
  bA0.u4[1] = *(const uint4*)(P0 + 8);
  bA1.u4[0] = *(const uint4*)(P1);
  bA1.u4[1] = *(const uint4*)(P1 + 8);
  for (int j0 = 0; j0 < Ntok; j0 += 64) {
    int j1 = j0 + 32;
    aB.u4[0]  = *(const uint4*)(Ebase + j1 + half * 16);
    aB.u4[1]  = *(const uint4*)(Ebase + j1 + half * 16 + 8);
    bB0.u4[0] = *(const uint4*)(P0 + j1);
    bB0.u4[1] = *(const uint4*)(P0 + j1 + 8);
    bB1.u4[0] = *(const uint4*)(P1 + j1);
    bB1.u4[1] = *(const uint4*)(P1 + j1 + 8);
    acc0 = wmma_bf16(aA.bf, bA0.bf, acc0);
    acc1 = wmma_bf16(aA.bf, bA1.bf, acc1);
    int j2 = (j0 + 64) & (Ntok - 1);         // wraps to 0 on last iter
    aA.u4[0]  = *(const uint4*)(Ebase + j2 + half * 16);
    aA.u4[1]  = *(const uint4*)(Ebase + j2 + half * 16 + 8);
    bA0.u4[0] = *(const uint4*)(P0 + j2);
    bA0.u4[1] = *(const uint4*)(P0 + j2 + 8);
    bA1.u4[0] = *(const uint4*)(P1 + j2);
    bA1.u4[1] = *(const uint4*)(P1 + j2 + 8);
    acc0 = wmma_bf16(aB.bf, bB0.bf, acc0);
    acc1 = wmma_bf16(aB.bf, bB1.bf, acc1);
  }

  // ---- stage pos_feat tile to LDS as bf16 (C layout: row = r + 8*half)
#pragma unroll
  for (int r = 0; r < 8; ++r) {
    int rowo = (wr * 16 + r + 8 * half) * Kdim;
    spf[rowo + wc * 32 + ln]      = (bf16)acc0[r];
    spf[rowo + wc * 32 + 16 + ln] = (bf16)acc1[r];
  }
  __syncthreads();

  // ---- GEMM3: out[i, d] += sum_k pf[i,k] * Wb[d,k]   (K = 128 = 4 chunks)
  V16 A[4];
  const bf16* lrow = spf + (wr * 16 + ln) * Kdim;
#pragma unroll
  for (int kc = 0; kc < 4; ++kc) {
    A[kc].u4[0] = *(const uint4*)(lrow + kc * 32 + half * 16);
    A[kc].u4[1] = *(const uint4*)(lrow + kc * 32 + half * 16 + 8);
  }

  V16 Bc[4], Bn[4];
  auto loadW = [&](int tile, V16* Bx) {
    const bf16* wrow =
        Wb + (size_t)(wc * 256 + tile * 16 + ln) * Kdim + half * 16;
#pragma unroll
    for (int kc = 0; kc < 4; ++kc) {
      Bx[kc].u4[0] = *(const uint4*)(wrow + kc * 32);
      Bx[kc].u4[1] = *(const uint4*)(wrow + kc * 32 + 8);
    }
  };
  auto tileOut = [&](int tile, const V16* Bx) {
    v8f c = {};
#pragma unroll
    for (int kc = 0; kc < 4; ++kc) c = wmma_bf16(A[kc].bf, Bx[kc].bf, c);
#pragma unroll
    for (int r = 0; r < 8; ++r) {
      int rl = wr * 16 + r + 8 * half;
      if (smask[rl] != 0) {  // padded rows: pos_feat forced to 0, keep base
        size_t o = ((size_t)(b * Ntok + i0 + rl)) * Dim + wc * 256 +
                   tile * 16 + ln;
        out[o] += c[r];
      }
    }
  };

  loadW(0, Bc);
#pragma unroll
  for (int tt = 0; tt < 16; tt += 2) {
    loadW(tt + 1, Bn);
    tileOut(tt, Bc);
    loadW((tt + 2) & 15, Bc);   // wraps on last iter (redundant, harmless)
    tileOut(tt + 1, Bn);
  }
}

// ---------------------------------------------------------------------------
extern "C" void kernel_launch(void* const* d_in, const int* in_sizes, int n_in,
                              void* d_out, int out_size, void* d_ws,
                              size_t ws_size, hipStream_t stream) {
  const int*   token = (const int*)d_in[0];
  const int*   attr  = (const int*)d_in[1];
  const float* pos   = (const float*)d_in[2];
  const float* embed = (const float*)d_in[3];
  const float* atom  = (const float*)d_in[4];
  const float* Wpos  = (const float*)d_in[5];
  const float* Wfeat = (const float*)d_in[6];
  float* out = (float*)d_out;

  char* ws = (char*)d_ws;
  bf16*  E     = (bf16*)(ws);                       // 33,554,432 B
  bf16*  Pt    = (bf16*)(ws + 33554432);            //  2,097,152 B
  bf16*  Wb    = (bf16*)(ws + 35651584);            //    262,144 B
  float* Zpart = (float*)(ws + 35913728);           //    262,144 B

  wconv_kernel<<<(Dim * Kdim) / 256, 256, 0, stream>>>(Wfeat, Wb);
  embed_kernel<<<Bsz * Ntok, 256, 0, stream>>>(token, attr, embed, atom, out);
  e_kernel<<<dim3(Ntok / 256, Ntok / 256, Bsz), 256, 0, stream>>>(token, pos, E,
                                                                  Zpart);
  p_kernel<<<dim3(Ntok / 256, Bsz), 256, 0, stream>>>(token, pos, Wpos, Zpart,
                                                      Pt);
  gemm_kernel<<<Bsz * (Ntok / 32), 256, 0, stream>>>(E, Pt, Wb, token, out);
}
